// minerva2_56135222559421
// MI455X (gfx1250) — matrix operand order repaired and checked by reading, not compiled, not run on hardware
//
#include <hip/hip_runtime.h>

typedef _Float16 f16;
typedef __attribute__((ext_vector_type(16))) _Float16 v16h;
typedef __attribute__((ext_vector_type(8)))  _Float16 v8h;
typedef __attribute__((ext_vector_type(8)))  float    v8f;

union V16 { v16h v; v8h p[2]; };

static constexpr int N_ROWS = 8192;
static constexpr int D_COLS = 16384;
static constexpr int H_DIM  = 256;
static constexpr int C_DIM  = 28;
static constexpr int CP     = 32;   // padded C: 28 data + 1 ones (intensity) + 3 zero
static constexpr int BM     = 128;  // N-rows per block
static constexpr int BD     = 128;  // D-cols per inner chunk
static constexpr int KC     = 32;   // WMMA K step (f16)
static constexpr int NSPLIT = 8;    // D splits -> grid.y
static constexpr int D_PER_SPLIT = D_COLS / NSPLIT; // 2048
static constexpr int KSTEPS = H_DIM / KC;           // 8

// ---------------------------------------------------------------------------
// CDNA5 async global->LDS copy (ASYNCcnt-tracked), per ISA 08_async_tensor §4.
// LDS operand is the wave-relative LDS byte address = low 32 bits of the
// generic pointer.
// ---------------------------------------------------------------------------
__device__ __forceinline__ uint32_t lds_off(const void* p) {
  return (uint32_t)(uintptr_t)p;
}
__device__ __forceinline__ void async_copy_b128(uint32_t lds, const void* g) {
  asm volatile("global_load_async_to_lds_b128 %0, %1, off"
               :: "v"(lds), "v"((unsigned long long)(uintptr_t)g) : "memory");
}
__device__ __forceinline__ void async_wait0() {
  asm volatile("s_wait_asynccnt 0" ::: "memory");
}

// ---------------------------------------------------------------------------
// Row L2-normalize (H=256) and convert to f16.  One block per row, 256 thr.
// ---------------------------------------------------------------------------
__global__ void rownorm_f16(const float* __restrict__ in, f16* __restrict__ out) {
  const int row = blockIdx.x;
  const int t   = threadIdx.x;                  // 0..255 == H
  const float x = in[(size_t)row * H_DIM + t];
  float s = x * x;
  #pragma unroll
  for (int off = 16; off > 0; off >>= 1) s += __shfl_xor(s, off, 32);
  __shared__ float red[8];
  if ((t & 31) == 0) red[t >> 5] = s;
  __syncthreads();
  float tot = 0.f;
  #pragma unroll
  for (int i = 0; i < 8; ++i) tot += red[i];
  const float inv = 1.0f / fmaxf(sqrtf(tot), 1e-12f);
  out[(size_t)row * H_DIM + t] = (f16)(x * inv);
}

// ---------------------------------------------------------------------------
// Rt[c][d] f16, c in [0,32): c<28 -> R[d][c]; c==28 -> 1 (intensity column);
// else 0.  d-major writes for coalescing.
// ---------------------------------------------------------------------------
__global__ void build_rt(const float* __restrict__ R, f16* __restrict__ Rt) {
  const int idx = blockIdx.x * blockDim.x + threadIdx.x;   // over CP*D_COLS
  const int c = idx / D_COLS;
  const int d = idx % D_COLS;
  float v = (c < C_DIM) ? R[(size_t)d * C_DIM + c] : (c == C_DIM ? 1.0f : 0.0f);
  Rt[idx] = (f16)v;
}

__global__ void zero_f32(float* __restrict__ p, int n) {
  const int i = blockIdx.x * blockDim.x + threadIdx.x;
  if (i < n) p[i] = 0.0f;
}

// ---------------------------------------------------------------------------
// Fused: s = Fn@En^T (f16 WMMA, f32 acc) -> a = s^3 -> echo/intensity via a
// second WMMA pass against Rt (ones column yields intensity).
// Staging copies use async global->LDS with explicit double buffering.
// ---------------------------------------------------------------------------
__launch_bounds__(256)
__global__ void minerva_fused(const f16* __restrict__ Fn, const f16* __restrict__ En,
                              const f16* __restrict__ Rt,
                              float* __restrict__ echo, float* __restrict__ inten) {
  __shared__ __align__(16) f16 sA[2][BM * KC];   // 16 KB  A k-slices (double buffered)
  __shared__ __align__(16) f16 sB[2][BD * KC];   // 16 KB  B k-slices (double buffered)
  __shared__ __align__(16) f16 sAt[BM * BD];     // 32 KB  a-tile (f16)
  f16* sR = &sB[1][0];   // Rt tile [CP*BD = 4096 halves] aliases sB[1] in echo phase

  const int t      = threadIdx.x;
  const int wave   = t >> 5;
  const int lane   = t & 31;
  const int laneHi = lane >> 4;     // 0: lanes 0-15, 1: lanes 16-31
  const int l16    = lane & 15;

  const int rowBase    = blockIdx.x * BM;
  const int dSplitBase = blockIdx.y * D_PER_SPLIT;

  // staging assignments: A/B: 2 threads/row, 32B (16 halves) each, as 2x16B chunks
  const int r2 = t >> 1, o2 = (t & 1) * 16;
  // Rt: 8 threads/row of 128 halves, 32B each
  const int c3 = t >> 3, o3 = (t & 7) * 16;

  // main-GEMM wave tile: 4x2 waves, each 32(M) x 64(N)
  const int wm = (wave & 3) * 32;
  const int wn = (wave >> 2) * 64;

  const v8f vzero = {0.f, 0.f, 0.f, 0.f, 0.f, 0.f, 0.f, 0.f};
  v8f eacc[2];                       // echo acc: rows 16*wave..+15, cols 0..31
  eacc[0] = vzero; eacc[1] = vzero;

  for (int dc = 0; dc < D_PER_SPLIT; dc += BD) {
    const int dBase = dSplitBase + dc;

    v8f acc[2][4];
    #pragma unroll
    for (int i = 0; i < 2; ++i)
      #pragma unroll
      for (int j = 0; j < 4; ++j) acc[i][j] = vzero;

    // prologue: async-issue k=0 slices into buffer 0
    {
      const f16* gA = &Fn[(size_t)(rowBase + r2) * H_DIM + o2];
      const f16* gB = &En[(size_t)(dBase  + r2) * H_DIM + o2];
      async_copy_b128(lds_off(&sA[0][r2 * KC + o2]),     gA);
      async_copy_b128(lds_off(&sA[0][r2 * KC + o2 + 8]), gA + 8);
      async_copy_b128(lds_off(&sB[0][r2 * KC + o2]),     gB);
      async_copy_b128(lds_off(&sB[0][r2 * KC + o2 + 8]), gB + 8);
    }

    // ---- main GEMM: K = H in steps of 32, async double-buffered ----
    for (int k = 0; k < KSTEPS; ++k) {
      const int cur = k & 1;
      async_wait0();        // this wave's step-k slices have landed in LDS
      __syncthreads();      // everyone's landed; everyone done reading buf[cur^1]
      if (k + 1 < KSTEPS) { // prefetch step k+1 into the other buffer
        const int nb = cur ^ 1, kn = (k + 1) * KC;
        const f16* gA = &Fn[(size_t)(rowBase + r2) * H_DIM + kn + o2];
        const f16* gB = &En[(size_t)(dBase  + r2) * H_DIM + kn + o2];
        async_copy_b128(lds_off(&sA[nb][r2 * KC + o2]),     gA);
        async_copy_b128(lds_off(&sA[nb][r2 * KC + o2 + 8]), gA + 8);
        async_copy_b128(lds_off(&sB[nb][r2 * KC + o2]),     gB);
        async_copy_b128(lds_off(&sB[nb][r2 * KC + o2 + 8]), gB + 8);
      }

      V16 af[2], bf[4];
      #pragma unroll
      for (int i = 0; i < 2; ++i) {
        // ISA 16-bit A layout: lanes<16 hold K 0..7 / 16..23, lanes>=16 hold 8..15 / 24..31
        const int m = wm + i * 16 + l16;
        af[i].p[0] = *(const v8h*)&sA[cur][m * KC + laneHi * 8];
        af[i].p[1] = *(const v8h*)&sA[cur][m * KC + 16 + laneHi * 8];
      }
      #pragma unroll
      for (int j = 0; j < 4; ++j) {
        // ISA 16-bit B layout: lane n holds col n, lanes<16 K 0..7 / 16..23, lanes>=16 K 8..15 / 24..31
        const int n = wn + j * 16 + l16;
        bf[j].p[0] = *(const v8h*)&sB[cur][n * KC + laneHi * 8];
        bf[j].p[1] = *(const v8h*)&sB[cur][n * KC + 16 + laneHi * 8];
      }
      #pragma unroll
      for (int i = 0; i < 2; ++i)
        #pragma unroll
        for (int j = 0; j < 4; ++j)
          acc[i][j] = __builtin_amdgcn_wmma_f32_16x16x32_f16(
              false, af[i].v, false, bf[j].v, (short)0, acc[i][j], false, false);
    }

    __syncthreads();   // all k=7 fragment reads done (sB[1] free for Rt alias)

    // ---- cube (a = s^3) -> a-tile f16 in LDS, row-major [m][d] ----
    // C/D layout: VGPR r -> M = r + 8*laneHi, N = l16
    #pragma unroll
    for (int i = 0; i < 2; ++i)
      #pragma unroll
      for (int j = 0; j < 4; ++j)
        #pragma unroll
        for (int r = 0; r < 8; ++r) {
          const float s = acc[i][j][r];
          const int m = wm + i * 16 + laneHi * 8 + r;
          const int n = wn + j * 16 + l16;
          sAt[m * BD + n] = (f16)(s * s * s);
        }

    // ---- async-load Rt tile [32][BD] into sR (aliases sB[1]) ----
    {
      const f16* gR = &Rt[(size_t)c3 * D_COLS + dBase + o3];
      async_copy_b128(lds_off(&sR[c3 * BD + o3]),     gR);
      async_copy_b128(lds_off(&sR[c3 * BD + o3 + 8]), gR + 8);
    }
    async_wait0();
    __syncthreads();

    // ---- echo GEMM: [128 x 32] += a_tile[128 x 128] @ RtT[128 x 32] ----
    #pragma unroll
    for (int k0 = 0; k0 < BD; k0 += KC) {
      V16 af;
      const int m = wave * 16 + l16;
      af.p[0] = *(const v8h*)&sAt[m * BD + k0 + laneHi * 8];
      af.p[1] = *(const v8h*)&sAt[m * BD + k0 + 16 + laneHi * 8];
      #pragma unroll
      for (int j = 0; j < 2; ++j) {
        V16 bf;
        const int c = j * 16 + l16;
        bf.p[0] = *(const v8h*)&sR[c * BD + k0 + laneHi * 8];
        bf.p[1] = *(const v8h*)&sR[c * BD + k0 + 16 + laneHi * 8];
        eacc[j] = __builtin_amdgcn_wmma_f32_16x16x32_f16(
            false, af.v, false, bf.v, (short)0, eacc[j], false, false);
      }
    }
  }

  // ---- flush echo/intensity partials (once per block: ~3.7K atomics) ----
  #pragma unroll
  for (int j = 0; j < 2; ++j)
    #pragma unroll
    for (int r = 0; r < 8; ++r) {
      const int m   = wave * 16 + laneHi * 8 + r;
      const int row = rowBase + m;
      const int c   = j * 16 + l16;
      const float v = eacc[j][r];
      if (c < C_DIM)            atomicAdd(&echo[(size_t)row * C_DIM + c], v);
      else if (c == C_DIM)      atomicAdd(&inten[row], v);
    }
}

// ---------------------------------------------------------------------------
extern "C" void kernel_launch(void* const* d_in, const int* in_sizes, int n_in,
                              void* d_out, int out_size, void* d_ws, size_t ws_size,
                              hipStream_t stream) {
  (void)in_sizes; (void)n_in; (void)out_size; (void)ws_size;
  const float* features      = (const float*)d_in[0];
  const float* ex_features   = (const float*)d_in[1];
  const float* ex_class_reps = (const float*)d_in[2];

  float* echo  = (float*)d_out;                       // [N, 28]
  float* inten = echo + (size_t)N_ROWS * C_DIM;       // [N]

  f16* Fn = (f16*)d_ws;                               // [N, H]   4 MB
  f16* En = Fn + (size_t)N_ROWS * H_DIM;              // [D, H]   8 MB
  f16* Rt = En + (size_t)D_COLS * H_DIM;              // [32, D]  1 MB

  rownorm_f16<<<N_ROWS, 256, 0, stream>>>(features, Fn);
  rownorm_f16<<<D_COLS, 256, 0, stream>>>(ex_features, En);
  build_rt<<<(CP * D_COLS) / 256, 256, 0, stream>>>(ex_class_reps, Rt);
  zero_f32<<<(N_ROWS * (C_DIM + 1) + 255) / 256, 256, 0, stream>>>(
      echo, N_ROWS * (C_DIM + 1));

  dim3 grid(N_ROWS / BM, NSPLIT);
  minerva_fused<<<grid, 256, 0, stream>>>(Fn, En, Rt, echo, inten);
}
